// RecurrentRGCN_6725918786050
// MI455X (gfx1250) — matrix operand order repaired and checked by reading, not compiled, not run on hardware
//
#include <hip/hip_runtime.h>
#include <hip/hip_bf16.h>
#include <math.h>

#define NE      50000
#define HD      256
#define R2      500
#define R2P     512
#define TSTEPS  4
#define NEDGE   200000
#define NM      200000
#define NL      2
#define SLOPE   0.22916666666666666f   // (1/8 + 1/3)/2

typedef __attribute__((ext_vector_type(16))) __bf16 v16bf;
typedef __attribute__((ext_vector_type(8)))  float  v8f;

struct __attribute__((aligned(16))) U4 { unsigned int x, y, z, w; };
union AF { v16bf v; U4 q[2]; };

__device__ inline float wave_sum(float v) {
  #pragma unroll
  for (int off = 16; off > 0; off >>= 1) v += __shfl_xor(v, off, 32);
  return v;
}

// ---- weight transpose+convert: src[k][n] f32 -> dst[n][k] bf16 -------------
__global__ void cvt_wT_kernel(const float* __restrict__ src, __hip_bfloat16* __restrict__ dst) {
  int idx = blockIdx.x * blockDim.x + threadIdx.x;   // HD*HD threads
  int k = idx >> 8, n = idx & (HD - 1);
  dst[n * HD + k] = __float2bfloat16(src[idx]);
}

// ---- f32 -> bf16 row copy with zero padding --------------------------------
__global__ void cvt_bf16_kernel(const float* __restrict__ in, __hip_bfloat16* __restrict__ out,
                                int n_valid, int n_total) {
  int idx = blockIdx.x * blockDim.x + threadIdx.x;
  if (idx >= n_total) return;
  out[idx] = __float2bfloat16(idx < n_valid ? in[idx] : 0.0f);
}

// ---- row-wise L2 normalize (one wave / row, 8 floats / lane) ---------------
__global__ void l2norm_kernel(const float* __restrict__ in, float* __restrict__ out, int rows) {
  int wid  = (blockIdx.x * blockDim.x + threadIdx.x) >> 5;
  int lane = threadIdx.x & 31;
  if (wid >= rows) return;
  const float* r = in + (size_t)wid * HD;
  float v[8]; float ss = 0.f;
  #pragma unroll
  for (int i = 0; i < 8; ++i) { v[i] = r[lane + 32 * i]; ss += v[i] * v[i]; }
  ss = wave_sum(ss);
  float inv = 1.0f / fmaxf(sqrtf(ss), 1e-12f);
  float* o = out + (size_t)wid * HD;
  #pragma unroll
  for (int i = 0; i < 8; ++i) o[lane + 32 * i] = v[i] * inv;
}

// ---- relation mean: scatter phase (one wave / entry) -----------------------
__global__ void relscat_kernel(const float* __restrict__ h, const int* __restrict__ ent,
                               const int* __restrict__ rel, float* __restrict__ xin,
                               float* __restrict__ cnt) {
  int w = blockIdx.x * 8 + (threadIdx.x >> 5);
  int lane = threadIdx.x & 31;
  if (w >= NM) return;
  int e = ent[w], r = rel[w];
  const float* s = h + (size_t)e * HD;
  float* d = xin + (size_t)r * HD;
  #pragma unroll
  for (int i = 0; i < 8; ++i) { int c = lane + 32 * i; atomicAdd(&d[c], s[c]); }
  if (lane == 0) atomicAdd(&cnt[r], 1.0f);
}

__global__ void reldiv_kernel(float* __restrict__ xin, const float* __restrict__ cnt) {
  int idx = blockIdx.x * blockDim.x + threadIdx.x;   // R2*HD
  xin[idx] /= fmaxf(cnt[idx >> 8], 1.0f);
}

// ---- GRU gate pre-activations: gx = [emb_rel|xin]@Wih^T+bih ; gh = h0@Whh^T+bhh
__global__ void gru_mm_kernel(const float* __restrict__ emb_rel, const float* __restrict__ xin,
                              const float* __restrict__ h0prev,
                              const float* __restrict__ wih, const float* __restrict__ whh,
                              const float* __restrict__ bih, const float* __restrict__ bhh,
                              float* __restrict__ gx, float* __restrict__ gh) {
  int idx = blockIdx.x * blockDim.x + threadIdx.x;
  if (idx >= R2 * 768) return;
  int row = idx / 768, o = idx % 768;
  const float* er = emb_rel + (size_t)row * HD;
  const float* xi = xin + (size_t)row * HD;
  const float* wr = wih + (size_t)o * 512;
  float s = bih[o];
  for (int k = 0; k < HD; ++k) s += er[k] * wr[k];
  for (int k = 0; k < HD; ++k) s += xi[k] * wr[HD + k];
  gx[idx] = s;
  const float* hr = h0prev + (size_t)row * HD;
  const float* wh = whh + (size_t)o * HD;
  float s2 = bhh[o];
  for (int k = 0; k < HD; ++k) s2 += hr[k] * wh[k];
  gh[idx] = s2;
}

// ---- GRU gates + L2 norm (one wave / relation row) -------------------------
__global__ void gru_gate_kernel(const float* __restrict__ gx, const float* __restrict__ gh,
                                const float* __restrict__ h0prev,
                                float* __restrict__ h0out, float* __restrict__ rels_out) {
  int row  = (blockIdx.x * blockDim.x + threadIdx.x) >> 5;
  int lane = threadIdx.x & 31;
  if (row >= R2) return;
  const float* gxr = gx + (size_t)row * 768;
  const float* ghr = gh + (size_t)row * 768;
  const float* hp  = h0prev + (size_t)row * HD;
  float val[8]; float ss = 0.f;
  #pragma unroll
  for (int i = 0; i < 8; ++i) {
    int c = lane + 32 * i;
    float r  = 1.0f / (1.0f + expf(-(gxr[c]       + ghr[c])));
    float z  = 1.0f / (1.0f + expf(-(gxr[256 + c] + ghr[256 + c])));
    float nn = tanhf(gxr[512 + c] + r * ghr[512 + c]);
    val[i] = (1.0f - z) * nn + z * hp[c];
    ss += val[i] * val[i];
  }
  ss = wave_sum(ss);
  float inv = 1.0f / fmaxf(sqrtf(ss), 1e-12f);
  float* ho = h0out + (size_t)row * HD;
  float* ro = rels_out + (size_t)row * HD;
  #pragma unroll
  for (int i = 0; i < 8; ++i) { int c = lane + 32 * i; float v = val[i] * inv; ho[c] = v; ro[c] = v; }
}

// ---- in-degree --------------------------------------------------------------
__global__ void deg_kernel(const int* __restrict__ dst, float* __restrict__ deg) {
  int e = blockIdx.x * blockDim.x + threadIdx.x;
  if (e >= NEDGE) return;
  atomicAdd(&deg[dst[e]], 1.0f);
}

// ---- fused 3-product WMMA GEMM: Hw = A@Wn ; Lsel = deg>0 ? A@Wl : A@We -----
// A: [NE][HD] bf16 row-major; BT_*: [HD][HD] bf16, [n][k] layout.
__global__ __launch_bounds__(256) void gemm3_kernel(
    const __hip_bfloat16* __restrict__ A_,
    const __hip_bfloat16* __restrict__ BTn_, const __hip_bfloat16* __restrict__ BTl_,
    const __hip_bfloat16* __restrict__ BTe_, const float* __restrict__ deg,
    float* __restrict__ Hw, float* __restrict__ Lsel) {
  const __bf16* A   = reinterpret_cast<const __bf16*>(A_);
  const __bf16* BTn = reinterpret_cast<const __bf16*>(BTn_);
  const __bf16* BTl = reinterpret_cast<const __bf16*>(BTl_);
  const __bf16* BTe = reinterpret_cast<const __bf16*>(BTe_);
  int g = blockIdx.x * 8 + (threadIdx.x >> 5);
  int lane = threadIdx.x & 31;
  int mt = g >> 4, nt = g & 15;
  if (mt >= NE / 16) return;
  int m0 = mt * 16, n0 = nt * 16;
  int half = lane >> 4, l16 = lane & 15;
  const __bf16* arow = A + (size_t)(m0 + l16) * HD;
  const __bf16* bn = BTn + (size_t)(n0 + l16) * HD + half * 16;
  const __bf16* bl = BTl + (size_t)(n0 + l16) * HD + half * 16;
  const __bf16* be = BTe + (size_t)(n0 + l16) * HD + half * 16;
  v8f accN = {0, 0, 0, 0, 0, 0, 0, 0};
  v8f accL = accN, accE = accN;
  #pragma unroll
  for (int k0 = 0; k0 < HD; k0 += 32) {
    AF a;
    a.q[0] = *(const U4*)(arow + k0 + half * 16);
    a.q[1] = *(const U4*)(arow + k0 + 8 + half * 16);
    v16bf vbn = *(const v16bf*)(bn + k0);
    v16bf vbl = *(const v16bf*)(bl + k0);
    v16bf vbe = *(const v16bf*)(be + k0);
    accN = __builtin_amdgcn_wmma_f32_16x16x32_bf16(false, a.v, false, vbn, (short)0, accN, false, false);
    accL = __builtin_amdgcn_wmma_f32_16x16x32_bf16(false, a.v, false, vbl, (short)0, accL, false, false);
    accE = __builtin_amdgcn_wmma_f32_16x16x32_bf16(false, a.v, false, vbe, (short)0, accE, false, false);
  }
  int n = n0 + l16;
  #pragma unroll
  for (int v = 0; v < 8; ++v) {
    int m = m0 + v + 8 * half;
    Hw[(size_t)m * HD + n]   = accN[v];
    Lsel[(size_t)m * HD + n] = (deg[m] > 0.f) ? accL[v] : accE[v];
  }
}

// ---- relation WMMA GEMM: Rw = h0 @ Wn (padded 512 rows) --------------------
__global__ __launch_bounds__(256) void gemm_rel_kernel(
    const __hip_bfloat16* __restrict__ A_, const __hip_bfloat16* __restrict__ BT_,
    float* __restrict__ Out) {
  const __bf16* A  = reinterpret_cast<const __bf16*>(A_);
  const __bf16* BT = reinterpret_cast<const __bf16*>(BT_);
  int g = blockIdx.x * 8 + (threadIdx.x >> 5);
  int lane = threadIdx.x & 31;
  int mt = g >> 4, nt = g & 15;
  if (mt >= R2P / 16) return;
  int m0 = mt * 16, n0 = nt * 16;
  int half = lane >> 4, l16 = lane & 15;
  const __bf16* arow = A + (size_t)(m0 + l16) * HD;
  const __bf16* b = BT + (size_t)(n0 + l16) * HD + half * 16;
  v8f acc = {0, 0, 0, 0, 0, 0, 0, 0};
  #pragma unroll
  for (int k0 = 0; k0 < HD; k0 += 32) {
    AF a;
    a.q[0] = *(const U4*)(arow + k0 + half * 16);
    a.q[1] = *(const U4*)(arow + k0 + 8 + half * 16);
    v16bf vb = *(const v16bf*)(b + k0);
    acc = __builtin_amdgcn_wmma_f32_16x16x32_bf16(false, a.v, false, vb, (short)0, acc, false, false);
  }
  int n = n0 + l16;
  #pragma unroll
  for (int v = 0; v < 8; ++v) Out[(size_t)(m0 + v + 8 * half) * HD + n] = acc[v];
}

// ---- edge pass: agg[dst] += Hw[src] + Rw[etype] (one wave / edge) ----------
__global__ __launch_bounds__(256) void edge_kernel(
    const int* __restrict__ src, const int* __restrict__ dst, const int* __restrict__ et,
    const float* __restrict__ Hw, const float* __restrict__ Rw, float* __restrict__ agg) {
  int e = blockIdx.x * 8 + (threadIdx.x >> 5);
  int lane = threadIdx.x & 31;
  if (e >= NEDGE) return;
  int s = src[e], d = dst[e], r = et[e];
  const float* hs = Hw + (size_t)s * HD;
  const float* rr = Rw + (size_t)r * HD;
  float* ad = agg + (size_t)d * HD;
  #pragma unroll
  for (int i = 0; i < 8; ++i) { int c = lane + 32 * i; atomicAdd(&ad[c], hs[c] + rr[c]); }
}

// ---- combine: hc = rrelu(agg / max(deg,1) + loop) --------------------------
__global__ void combine_kernel(const float* __restrict__ agg, const float* __restrict__ deg,
                               const float* __restrict__ lsel, float* __restrict__ hc) {
  int idx = blockIdx.x * blockDim.x + threadIdx.x;   // NE*HD
  float nr = 1.0f / fmaxf(deg[idx >> 8], 1.0f);
  float v = agg[idx] * nr + lsel[idx];
  hc[idx] = (v >= 0.f) ? v : SLOPE * v;
}

extern "C" void kernel_launch(void* const* d_in, const int* in_sizes, int n_in,
                              void* d_out, int out_size, void* d_ws, size_t ws_size,
                              hipStream_t stream) {
  const int*   src     = (const int*)d_in[0];
  const int*   dst     = (const int*)d_in[1];
  const int*   etype   = (const int*)d_in[2];
  const int*   r2e_ent = (const int*)d_in[3];
  const int*   r2e_rel = (const int*)d_in[4];
  const float* dyn     = (const float*)d_in[5];
  const float* emb_rel = (const float*)d_in[6];
  const float* w_nb    = (const float*)d_in[7];
  const float* w_lp    = (const float*)d_in[8];
  const float* w_ev    = (const float*)d_in[9];
  const float* w_ih    = (const float*)d_in[10];
  const float* w_hh    = (const float*)d_in[11];
  const float* b_ih    = (const float*)d_in[12];
  const float* b_hh    = (const float*)d_in[13];

  char* p = (char*)d_ws;
  auto carve = [&](size_t bytes) -> void* {
    void* q = (void*)p; p += (bytes + 255) & ~(size_t)255; return q;
  };
  float* hc   = (float*)carve((size_t)NE * HD * 4);
  float* Hw   = (float*)carve((size_t)NE * HD * 4);
  float* Lsel = (float*)carve((size_t)NE * HD * 4);
  float* agg  = (float*)carve((size_t)NE * HD * 4);
  __hip_bfloat16* hcb = (__hip_bfloat16*)carve((size_t)NE * HD * 2);
  float* deg  = (float*)carve((size_t)NE * 4);
  float* h0   = (float*)carve((size_t)R2P * HD * 4);
  __hip_bfloat16* h0b = (__hip_bfloat16*)carve((size_t)R2P * HD * 2);
  float* Rw   = (float*)carve((size_t)R2P * HD * 4);
  float* xin  = (float*)carve((size_t)R2 * HD * 4);
  float* cnt  = (float*)carve((size_t)R2P * 4);
  float* gx   = (float*)carve((size_t)R2 * 768 * 4);
  float* gh   = (float*)carve((size_t)R2 * 768 * 4);
  __hip_bfloat16* wnT = (__hip_bfloat16*)carve((size_t)NL * HD * HD * 2);
  __hip_bfloat16* lpT = (__hip_bfloat16*)carve((size_t)NL * HD * HD * 2);
  __hip_bfloat16* evT = (__hip_bfloat16*)carve((size_t)NL * HD * HD * 2);

  float* hist = (float*)d_out;
  float* rels = hist + (size_t)TSTEPS * NE * HD;

  // fragment-friendly bf16 weight layouts ([n][k])
  for (int l = 0; l < NL; ++l) {
    cvt_wT_kernel<<<HD * HD / 256, 256, 0, stream>>>(w_nb + (size_t)l * HD * HD, wnT + (size_t)l * HD * HD);
    cvt_wT_kernel<<<HD * HD / 256, 256, 0, stream>>>(w_lp + (size_t)l * HD * HD, lpT + (size_t)l * HD * HD);
    cvt_wT_kernel<<<HD * HD / 256, 256, 0, stream>>>(w_ev + (size_t)l * HD * HD, evT + (size_t)l * HD * HD);
  }
  // h = l2norm(dynamic_emb)
  l2norm_kernel<<<(NE + 7) / 8, 256, 0, stream>>>(dyn, hc, NE);

  for (int t = 0; t < TSTEPS; ++t) {
    const float* h_in   = (t == 0) ? hc : (hist + (size_t)(t - 1) * NE * HD);
    const float* h0prev = (t == 0) ? emb_rel : h0;

    // relation-wise mean of entity embeddings
    hipMemsetAsync(xin, 0, (size_t)R2 * HD * 4, stream);
    hipMemsetAsync(cnt, 0, (size_t)R2P * 4, stream);
    relscat_kernel<<<(NM + 7) / 8, 256, 0, stream>>>(h_in, r2e_ent + (size_t)t * NM,
                                                     r2e_rel + (size_t)t * NM, xin, cnt);
    reldiv_kernel<<<R2 * HD / 256, 256, 0, stream>>>(xin, cnt);

    // GRUCell update of relation embeddings
    gru_mm_kernel<<<(R2 * 768 + 255) / 256, 256, 0, stream>>>(emb_rel, xin, h0prev,
                                                              w_ih, w_hh, b_ih, b_hh, gx, gh);
    gru_gate_kernel<<<(R2 + 7) / 8, 256, 0, stream>>>(gx, gh, h0prev, h0,
                                                      rels + (size_t)t * R2 * HD);
    cvt_bf16_kernel<<<(R2P * HD + 255) / 256, 256, 0, stream>>>(h0, h0b, R2 * HD, R2P * HD);

    // in-degree (per step, shared by both layers)
    hipMemsetAsync(deg, 0, (size_t)NE * 4, stream);
    deg_kernel<<<(NEDGE + 255) / 256, 256, 0, stream>>>(dst + (size_t)t * NEDGE, deg);

    for (int l = 0; l < NL; ++l) {
      const float* cur = (l == 0) ? h_in : hc;
      cvt_bf16_kernel<<<((size_t)NE * HD + 255) / 256, 256, 0, stream>>>(cur, hcb, NE * HD, NE * HD);
      gemm_rel_kernel<<<R2P * 16 / 16 / 8, 256, 0, stream>>>(h0b, wnT + (size_t)l * HD * HD, Rw);
      gemm3_kernel<<<NE * 16 / 16 / 8, 256, 0, stream>>>(hcb, wnT + (size_t)l * HD * HD,
                                                         lpT + (size_t)l * HD * HD,
                                                         evT + (size_t)l * HD * HD, deg, Hw, Lsel);
      hipMemsetAsync(agg, 0, (size_t)NE * HD * 4, stream);
      edge_kernel<<<(NEDGE + 7) / 8, 256, 0, stream>>>(src + (size_t)t * NEDGE,
                                                       dst + (size_t)t * NEDGE,
                                                       etype + (size_t)t * NEDGE, Hw, Rw, agg);
      combine_kernel<<<NE * HD / 256, 256, 0, stream>>>(agg, deg, Lsel, hc);
    }
    l2norm_kernel<<<(NE + 7) / 8, 256, 0, stream>>>(hc, hist + (size_t)t * NE * HD, NE);
  }
}
